// StackedLSTMCell_76012331204660
// MI455X (gfx1250) — matrix operand order repaired and checked by reading, not compiled, not run on hardware
//
#include <hip/hip_runtime.h>
#include <cmath>

// ---------------------------------------------------------------------------
// Fused StackedLSTM layer for MI455X (gfx1250, wave32).
//   Z = X@W + H@U + b  via V_WMMA_F32_16X16X32_F16 (f16 inputs, f32 accum)
//   c = sig(zf)*c_tm1 + sig(zi)*tanh(zg);  h = sig(zo)*tanh(c)
// NNI naming swap: GEMM recurrent input = prev_c[l], elementwise c_tm1 = prev_h[l].
// ---------------------------------------------------------------------------

typedef _Float16 h1;
typedef __attribute__((ext_vector_type(4)))  _Float16 h4;
typedef __attribute__((ext_vector_type(8)))  _Float16 h8;
typedef __attribute__((ext_vector_type(16))) _Float16 v16h;
typedef __attribute__((ext_vector_type(4)))  float    f4;
typedef __attribute__((ext_vector_type(8)))  float    v8f;

constexpr int BATCH = 16384;
constexpr int S     = 512;
constexpr int FS    = 2048;    // 4*S

constexpr int BM  = 128;       // rows per block
constexpr int BK  = 32;        // K per stage == K per WMMA
constexpr int SA  = 40;        // LDS stride (halfs) for A and Bt tiles: 80B rows,
                               // 16B-aligned b128 frag loads, conflict-free banks
constexpr int LDZ = 128;       // Z spill stride (f32); smem = 128*128*4 = 64KB

__global__ __launch_bounds__(256, 2)
void lstm_layer_kernel(const float* __restrict__ X,      // [B,S] layer input
                       const float* __restrict__ Hrec,   // [B,S] recurrent input (= prev_c[l])
                       const float* __restrict__ Cprev,  // [B,S] cell state     (= prev_h[l])
                       const float* __restrict__ W,      // [S,4S]
                       const float* __restrict__ U,      // [S,4S]
                       const float* __restrict__ bias,   // [4S]
                       float* __restrict__ Hout,         // [B,S]
                       float* __restrict__ Cout)         // [B,S]
{
    __shared__ float smem[BM * LDZ];          // 64KB; staging region aliases Z region
    h1* Ah = (h1*)smem;                       // A tile: 128 rows x SA halfs ([m][k])
    h1* Bh = Ah + BM * SA;                    // B tile: 128 rows x SA halfs, transposed ([n][k])
    float* Zs = smem;                         // Z spill after GEMM loop

    const int t    = threadIdx.x;
    const int lane = t & 31;
    const int wave = t >> 5;
    const int m0   = blockIdx.y * BM;         // row base
    const int n0   = blockIdx.x * 32;         // per-gate column base

    // 8 waves tile 128x128: rows [wm*32,+32), local cols [wn*64,+64)
    const int wm = wave & 3;
    const int wn = wave >> 2;

    v8f acc[2][4] = {};                       // 2 x 4 tiles of 16x16 f32

    // ---- staging maps -----------------------------------------------------
    // A: thread -> one row, 16 of the 32 k's (4 coalesced float4 loads)
    const int a_r  = t >> 1;                  // 0..127
    const int a_k0 = (t & 1) * 16;            // 0 | 16
    // B: thread -> 4x4 sub-block (4 coalesced float4 row loads, reg transpose,
    //    4 contiguous b64 stores into the [n][k] tile)
    const int b_ng   = t & 31;                // n-group 0..31
    const int b_kg   = t >> 5;                // k-group 0..7
    const int b_lc   = b_ng * 4;              // local col 0..124 (within one gate strip)
    const int b_gcol = (b_lc >> 5) * S + n0 + (b_lc & 31);   // gathered gate column

    f4 aR[4], bR[4];

    auto loadGlobal = [&](int k0) {
        const float* Asrc; const float* Bsrc; int kk;
        if (k0 < S) { Asrc = X;    Bsrc = W; kk = k0;     }
        else        { Asrc = Hrec; Bsrc = U; kk = k0 - S; }
        const float* ap = Asrc + (size_t)(m0 + a_r) * S + kk + a_k0;
        aR[0] = *(const f4*)(ap);
        aR[1] = *(const f4*)(ap + 4);
        aR[2] = *(const f4*)(ap + 8);
        aR[3] = *(const f4*)(ap + 12);
        const float* bp = Bsrc + (size_t)(kk + b_kg * 4) * FS + b_gcol;
        bR[0] = *(const f4*)(bp);
        bR[1] = *(const f4*)(bp + FS);
        bR[2] = *(const f4*)(bp + 2 * FS);
        bR[3] = *(const f4*)(bp + 3 * FS);
    };

    auto storeLDS = [&]() {
        h8 p0 = { (h1)aR[0][0], (h1)aR[0][1], (h1)aR[0][2], (h1)aR[0][3],
                  (h1)aR[1][0], (h1)aR[1][1], (h1)aR[1][2], (h1)aR[1][3] };
        h8 p1 = { (h1)aR[2][0], (h1)aR[2][1], (h1)aR[2][2], (h1)aR[2][3],
                  (h1)aR[3][0], (h1)aR[3][1], (h1)aR[3][2], (h1)aR[3][3] };
        *(h8*)(Ah + a_r * SA + a_k0)     = p0;
        *(h8*)(Ah + a_r * SA + a_k0 + 8) = p1;
        #pragma unroll
        for (int j = 0; j < 4; ++j) {
            h4 p = { (h1)bR[0][j], (h1)bR[1][j], (h1)bR[2][j], (h1)bR[3][j] };
            *(h4*)(Bh + (b_lc + j) * SA + b_kg * 4) = p;
        }
    };

    // Fragment layouts used here (16-bit, wave32, contiguous k, same order for A and B):
    //  A 16x32: lanes 0-15 (M=lane): K{0..15}; lanes 16-31: K{16..31}
    //  B 32x16: lanes 0-15 (N=lane): K{0..15}; lanes 16-31: K{16..31}
    const int l16 = lane & 15;
    const int kh  = lane >> 4;

    auto compute = [&]() {
        v16h afr[2], bfr[4];
        #pragma unroll
        for (int i = 0; i < 2; ++i) {
            const h1* p = Ah + (wm * 32 + i * 16 + l16) * SA;
            h8 lo = *(const h8*)(p + 16 * kh);
            h8 hi = *(const h8*)(p + 16 * kh + 8);
            afr[i] = __builtin_shufflevector(lo, hi,
                        0,1,2,3,4,5,6,7,8,9,10,11,12,13,14,15);
        }
        #pragma unroll
        for (int j = 0; j < 4; ++j) {
            const h1* p = Bh + (wn * 64 + j * 16 + l16) * SA + 16 * kh;
            h8 lo = *(const h8*)(p);
            h8 hi = *(const h8*)(p + 8);
            bfr[j] = __builtin_shufflevector(lo, hi,
                        0,1,2,3,4,5,6,7,8,9,10,11,12,13,14,15);
        }
        #pragma unroll
        for (int j = 0; j < 4; ++j)
            #pragma unroll
            for (int i = 0; i < 2; ++i)
                acc[i][j] = __builtin_amdgcn_wmma_f32_16x16x32_f16(
                    false, afr[i], false, bfr[j], (short)0, acc[i][j],
                    false, false);
    };

    // ---- software-pipelined K loop over both GEMMs (K_total = 2*S = 1024) ----
    loadGlobal(0);
    for (int k0 = 0; k0 < 2 * S; k0 += BK) {
        storeLDS();
        __syncthreads();
        if (k0 + BK < 2 * S) loadGlobal(k0 + BK);
        compute();
        __syncthreads();
    }

    // ---- spill accumulators to LDS so gates can be recombined across waves ----
    #pragma unroll
    for (int i = 0; i < 2; ++i)
        #pragma unroll
        for (int j = 0; j < 4; ++j) {
            const int colb = wn * 64 + j * 16 + l16;
            const int rowb = wm * 32 + i * 16 + kh * 8;
            #pragma unroll
            for (int v = 0; v < 8; ++v)
                Zs[(rowb + v) * LDZ + colb] = acc[i][j][v];
        }
    __syncthreads();

    // ---- fused gate nonlinearity + outputs (128 rows x 32 cols per block) ----
    const int jj   = t & 31;
    const int gcol = n0 + jj;
    const float bi = bias[gcol];
    const float bf = bias[S + gcol];
    const float bg = bias[2 * S + gcol];
    const float bo = bias[3 * S + gcol];
    const int rbase = (t >> 5) * 16;
    #pragma unroll
    for (int rr = 0; rr < 16; ++rr) {
        const int r = rbase + rr;
        const float zi = Zs[r * LDZ + jj]      + bi;
        const float zf = Zs[r * LDZ + 32 + jj] + bf;
        const float zg = Zs[r * LDZ + 64 + jj] + bg;
        const float zo = Zs[r * LDZ + 96 + jj] + bo;
        const size_t gidx = (size_t)(m0 + r) * S + gcol;
        const float cprev = Cprev[gidx];
        const float si = 1.f / (1.f + __expf(-zi));
        const float sf = 1.f / (1.f + __expf(-zf));
        const float so = 1.f / (1.f + __expf(-zo));
        const float c  = sf * cprev + si * tanhf(zg);
        const float h  = so * tanhf(c);
        Cout[gidx] = c;
        Hout[gidx] = h;
    }
}

extern "C" void kernel_launch(void* const* d_in, const int* in_sizes, int n_in,
                              void* d_out, int out_size, void* d_ws, size_t ws_size,
                              hipStream_t stream) {
    (void)in_sizes; (void)n_in; (void)out_size; (void)d_ws; (void)ws_size;
    const float* inputs   = (const float*)d_in[0];
    const float* prev_c   = (const float*)d_in[1];
    const float* prev_h   = (const float*)d_in[2];
    const float* kernels  = (const float*)d_in[3];
    const float* rec_kern = (const float*)d_in[4];
    const float* biases   = (const float*)d_in[5];
    float* out = (float*)d_out;

    const size_t BS = (size_t)BATCH * S;
    dim3 grid(S / 32, BATCH / BM);   // 16 x 128 = 2048 workgroups
    dim3 block(256);

    for (int l = 0; l < 2; ++l) {
        // layer input: layer0 = inputs; layer1 = c of layer0 (out[0,1,:,:])
        const float* x = (l == 0) ? inputs : out + BS;
        hipLaunchKernelGGL(lstm_layer_kernel, grid, block, 0, stream,
                           x,
                           prev_c + (size_t)l * BS,          // GEMM recurrent input (NNI swap)
                           prev_h + (size_t)l * BS,          // elementwise c_tm1    (NNI swap)
                           kernels  + (size_t)l * S * FS,
                           rec_kern + (size_t)l * S * FS,
                           biases   + (size_t)l * FS,
                           out + (size_t)(2 * l)     * BS,   // h  -> out[l,0]
                           out + (size_t)(2 * l + 1) * BS);  // c  -> out[l,1]
    }
}
